// WorldModel_86466281603575
// MI455X (gfx1250) — matrix operand order repaired and checked by reading, not compiled, not run on hardware
//
#include <hip/hip_runtime.h>
#include <hip/hip_bf16.h>
#include <stdint.h>

// ---------------------------------------------------------------------------
// WorldModel forward for MI455X (gfx1250, wave32, WMMA).
// All heavy math runs through v_wmma_f32_16x16x32_bf16 (bf16 in, f32 acc).
// Wave tile: 32(M) x 64(N) -> 8 WMMAs per 32-wide K step.
// ---------------------------------------------------------------------------

typedef __bf16 bf16;
typedef __attribute__((ext_vector_type(16))) __bf16 v16bf;
typedef __attribute__((ext_vector_type(8)))  float  v8f;

#define T_STEPS 63      // T-1
#define BATCH   256
#define OBS_D   512
#define ACT_D   32
#define EMB_D   1024
#define DET_D   1024
#define STO_D   256
#define NROWS   (T_STEPS * BATCH)   // 16128

// ---------------- helpers ----------------

__device__ __forceinline__ float block_reduce_sum(float v) {
    __shared__ float s[256];
    int t = threadIdx.x;
    __syncthreads();              // guard against back-to-back invocations
    s[t] = v;
    __syncthreads();
    for (int o = 128; o > 0; o >>= 1) {
        if (t < o) s[t] += s[t + o];
        __syncthreads();
    }
    return s[0];
}

__device__ __forceinline__ v16bf load_b_frag(const bf16* p) {
    // 16 contiguous bf16 (32 bytes)
    union { v16bf v; uint4 u[2]; } x;
    x.u[0] = ((const uint4*)p)[0];
    x.u[1] = ((const uint4*)p)[1];
    return x.v;
}

__device__ __forceinline__ v16bf load_a_frag(const bf16* p) {
    // 16 contiguous bf16: elements 0..15 = K+0..15
    union { v16bf v; uint4 u[2]; } x;
    x.u[0] = ((const uint4*)p)[0];
    x.u[1] = ((const uint4*)(p + 8))[0];
    return x.v;
}

// Inner K loop over one A source. a0p/a1p: row base (+klo) for the two 16-row
// A fragments; bp: B base (+kb) for this wave's 64-col strip; ldbK = K.
__device__ __forceinline__ void gemm_k_loop(const bf16* __restrict__ a0p,
                                            const bf16* __restrict__ a1p,
                                            const bf16* __restrict__ bp,
                                            int kcount, size_t ldbK,
                                            v8f (&acc)[2][4])
{
    for (int k = 0; k < kcount; k += 32) {
        v16bf a0 = load_a_frag(a0p + k);
        v16bf a1 = load_a_frag(a1p + k);
        const bf16* brow = bp + k;
        v16bf b0 = load_b_frag(brow);
        v16bf b1 = load_b_frag(brow + 16 * ldbK);
        v16bf b2 = load_b_frag(brow + 32 * ldbK);
        v16bf b3 = load_b_frag(brow + 48 * ldbK);
        // prefetch B ahead (L2-resident working set -> pull into near cache)
        __builtin_prefetch(brow + 256, 0, 1);

        acc[0][0] = __builtin_amdgcn_wmma_f32_16x16x32_bf16(false, a0, false, b0, (short)0, acc[0][0], false, false);
        acc[0][1] = __builtin_amdgcn_wmma_f32_16x16x32_bf16(false, a0, false, b1, (short)0, acc[0][1], false, false);
        acc[0][2] = __builtin_amdgcn_wmma_f32_16x16x32_bf16(false, a0, false, b2, (short)0, acc[0][2], false, false);
        acc[0][3] = __builtin_amdgcn_wmma_f32_16x16x32_bf16(false, a0, false, b3, (short)0, acc[0][3], false, false);
        acc[1][0] = __builtin_amdgcn_wmma_f32_16x16x32_bf16(false, a1, false, b0, (short)0, acc[1][0], false, false);
        acc[1][1] = __builtin_amdgcn_wmma_f32_16x16x32_bf16(false, a1, false, b1, (short)0, acc[1][1], false, false);
        acc[1][2] = __builtin_amdgcn_wmma_f32_16x16x32_bf16(false, a1, false, b2, (short)0, acc[1][2], false, false);
        acc[1][3] = __builtin_amdgcn_wmma_f32_16x16x32_bf16(false, a1, false, b3, (short)0, acc[1][3], false, false);
    }
}

// ---------------- generic bf16 WMMA GEMM ----------------
// C(MxN) = act( [A0|A1](MxK) * BT(NxK)^T + bias ),  optional f32 / bf16 out.
// Split-A: k < ksplit reads A0 (lda0), else A1 (lda1) at k-ksplit; the split
// is handled as two separate K loops (no per-iteration selects).
__global__ __launch_bounds__(256)
void gemm_bf16_kernel(const bf16* __restrict__ A0, int lda0,
                      const bf16* __restrict__ A1, int lda1, int ksplit,
                      const bf16* __restrict__ BT,
                      const float* __restrict__ bias,
                      float* __restrict__ Cf, bf16* __restrict__ Cb,
                      int M, int N, int K, int ldc, int act)
{
    const int wave = (blockIdx.x * blockDim.x + threadIdx.x) >> 5;
    const int lane = threadIdx.x & 31;
    const int ntiles = N >> 6;                 // 64-wide strips
    const int mt = wave / ntiles;              // 32-row tiles
    const int nt = wave % ntiles;
    if (mt >= (M >> 5)) return;

    const int mrow0 = mt * 32 + (lane & 15);   // A frag 0: rows 0..15 of tile
    const int mrow1 = mrow0 + 16;              // A frag 1: rows 16..31
    const int klo   = (lane >> 4) << 4;        // A: 0 or 16
    const int kb    = (lane >> 4) << 4;        // B: 0 or 16
    const int ncol  = nt * 64 + (lane & 15);

    v8f acc[2][4] = {};

    const bf16* bp = BT + (size_t)ncol * K + kb;
    const int k0 = (ksplit < K) ? ksplit : K;

    // segment 1: A0
    gemm_k_loop(A0 + (size_t)mrow0 * lda0 + klo,
                A0 + (size_t)mrow1 * lda0 + klo,
                bp, k0, (size_t)K, acc);
    // segment 2: A1 (concatenated along K)
    if (K > k0) {
        gemm_k_loop(A1 + (size_t)mrow0 * lda1 + klo,
                    A1 + (size_t)mrow1 * lda1 + klo,
                    bp + k0, K - k0, (size_t)K, acc);
    }

    // C/D layout: VGPR r, lane L -> row half*16 + r + 8*(L>=16), col = (L&15)
    const int rlo = (lane >> 4) << 3;
#pragma unroll
    for (int half = 0; half < 2; ++half) {
#pragma unroll
        for (int sub = 0; sub < 4; ++sub) {
            int col = nt * 64 + sub * 16 + (lane & 15);
            float bv = bias ? bias[col] : 0.0f;
#pragma unroll
            for (int r = 0; r < 8; ++r) {
                int row = mt * 32 + half * 16 + rlo + r;
                if (row < M) {
                    float v = acc[half][sub][r] + bv;
                    if (act) v = fmaxf(v, 0.0f);
                    if (Cf) Cf[(size_t)row * ldc + col] = v;
                    if (Cb) Cb[(size_t)row * ldc + col] = (bf16)v;
                }
            }
        }
    }
}

// ---------------- packing / conversion ----------------

__global__ void pack_transpose_kernel(const float* __restrict__ W, bf16* __restrict__ WT,
                                      int K, int N) {  // W: KxN -> WT: NxK
    int idx = blockIdx.x * blockDim.x + threadIdx.x;
    if (idx >= K * N) return;
    int n = idx / K, k = idx % K;
    WT[(size_t)n * K + k] = (bf16)W[(size_t)k * N + n];
}

__global__ void pack_convert_kernel(const float* __restrict__ W, bf16* __restrict__ WT, int n) {
    int i = blockIdx.x * blockDim.x + threadIdx.x;
    if (i < n) WT[i] = (bf16)W[i];
}

__global__ void f32_to_bf16_kernel(const float* __restrict__ src, bf16* __restrict__ dst, int n) {
    int i = blockIdx.x * blockDim.x + threadIdx.x;
    if (i < n) dst[i] = (bf16)src[i];
}

__global__ void zero_h_kernel(float* __restrict__ hf, bf16* __restrict__ hb) {
    int i = blockIdx.x * blockDim.x + threadIdx.x;   // 256*1024
    hf[i] = 0.0f; hb[i] = (bf16)0.0f;
}

// ---------------- per-step elementwise kernels ----------------

// z = qm + eps*exp(ql); KL per row. grid=(B), block=(STO)
__global__ void latent_kl_kernel(const float* __restrict__ qq, const float* __restrict__ pp,
                                 const float* __restrict__ eps_t,
                                 bf16* __restrict__ z_out, float* __restrict__ kl_out)
{
    int b = blockIdx.x, j = threadIdx.x;
    float qm = qq[b * 512 + j];
    float ql = qq[b * 512 + 256 + j];
    float pm = pp[b * 512 + j];
    float pl = pp[b * 512 + 256 + j];
    float e  = eps_t[b * 256 + j];
    float z  = qm + e * expf(ql);
    z_out[b * 256 + j] = (bf16)z;
    float vq = expf(2.0f * ql), vp = expf(2.0f * pl);
    float d  = qm - pm;
    float kle = pl - ql + (vq + d * d) / (vp + 1e-8f) - 1.0f;
    float s = block_reduce_sum(kle);
    if (j == 0) kl_out[b] = 0.5f * s;
}

// GRU gate math + state update. grid covers B*DET.
__global__ void gru_update_kernel(const float* __restrict__ gi, const float* __restrict__ gh,
                                  float* __restrict__ h_f32, bf16* __restrict__ h_bf,
                                  bf16* __restrict__ hseq_t,
                                  const unsigned char* __restrict__ dones_t)
{
    int idx = blockIdx.x * blockDim.x + threadIdx.x;
    int b = idx >> 10, j = idx & 1023;
    const float* gib = gi + (size_t)b * 3072;
    const float* ghb = gh + (size_t)b * 3072;
    float r = 1.0f / (1.0f + expf(-(gib[j]        + ghb[j])));
    float u = 1.0f / (1.0f + expf(-(gib[1024 + j] + ghb[1024 + j])));
    float n = tanhf(gib[2048 + j] + r * ghb[2048 + j]);
    float hprev = h_f32[idx];
    float hnext = (1.0f - u) * n + u * hprev;
    hseq_t[idx] = (bf16)hnext;                      // unmasked, for heads
    float mask = dones_t[b] ? 0.0f : 1.0f;
    float hm = hnext * mask;                        // masked carry
    h_f32[idx] = hm;
    h_bf[idx]  = (bf16)hm;
}

// ---------------- post-scan loss kernels ----------------

__global__ void rew2_loss_kernel(const bf16* __restrict__ r1, const float* __restrict__ w2,
                                 const float* __restrict__ b2, const float* __restrict__ rewards,
                                 float* __restrict__ rewl)
{
    int i = blockIdx.x, t = threadIdx.x;
    float s = 0.0f;
    for (int k = t; k < 1024; k += 256) s += (float)r1[(size_t)i * 1024 + k] * w2[k];
    s = block_reduce_sum(s);
    if (t == 0) { float p = s + b2[0]; float d = p - rewards[i]; rewl[i] = d * d; }
}

__global__ void recon_kernel(const float* __restrict__ pred, const float* __restrict__ obs,
                             float* __restrict__ recon)
{
    int i = blockIdx.x, t = threadIdx.x;
    float s = 0.0f;
    for (int c = t; c < 512; c += 256) {
        float d = pred[(size_t)i * 512 + c] - obs[(size_t)(i + BATCH) * 512 + c]; // obs[t+1]
        s += d * d;
    }
    s = block_reduce_sum(s);
    if (t == 0) recon[i] = s * (1.0f / 512.0f);
}

__global__ void final_reduce_kernel(const float* __restrict__ recon, const float* __restrict__ rewl,
                                    const float* __restrict__ kl, float* __restrict__ out, int n)
{
    float sr = 0, sw = 0, sk = 0;
    for (int i = threadIdx.x; i < n; i += 256) { sr += recon[i]; sw += rewl[i]; sk += kl[i]; }
    sr = block_reduce_sum(sr);
    sw = block_reduce_sum(sw);
    sk = block_reduce_sum(sk);
    if (threadIdx.x == 0) {
        float inv = 1.0f / (float)n;
        float r = sr * inv, w = sw * inv, k = sk * inv;
        out[0] = r + w + k; out[1] = r; out[2] = w; out[3] = k;
    }
}

// ---------------- host side ----------------

static inline void launch_gemm(hipStream_t s,
                               const bf16* A0, int lda0, const bf16* A1, int lda1, int ksplit,
                               const bf16* BT, const float* bias,
                               float* Cf, bf16* Cb, int M, int N, int K, int act)
{
    int waves  = (M / 32) * (N / 64);
    int blocks = (waves + 7) / 8;
    gemm_bf16_kernel<<<blocks, 256, 0, s>>>(A0, lda0, A1, lda1, ksplit, BT, bias,
                                            Cf, Cb, M, N, K, N, act);
}

extern "C" void kernel_launch(void* const* d_in, const int* in_sizes, int n_in,
                              void* d_out, int out_size, void* d_ws, size_t ws_size,
                              hipStream_t stream)
{
    (void)in_sizes; (void)n_in; (void)out_size; (void)ws_size;

    const float* obs     = (const float*)d_in[0];
    const float* actions = (const float*)d_in[1];
    const float* rewards = (const float*)d_in[2];
    const unsigned char* dones = (const unsigned char*)d_in[3];
    const float* eps     = (const float*)d_in[4];
    const float* enc_w1  = (const float*)d_in[5];
    const float* enc_b1  = (const float*)d_in[6];
    const float* enc_w2  = (const float*)d_in[7];
    const float* enc_b2  = (const float*)d_in[8];
    const float* gru_wih = (const float*)d_in[9];
    const float* gru_whh = (const float*)d_in[10];
    const float* gru_bih = (const float*)d_in[11];
    const float* gru_bhh = (const float*)d_in[12];
    const float* prior_w = (const float*)d_in[13];
    const float* prior_b = (const float*)d_in[14];
    const float* post_w  = (const float*)d_in[15];
    const float* post_b  = (const float*)d_in[16];
    const float* dec_w1  = (const float*)d_in[17];
    const float* dec_b1  = (const float*)d_in[18];
    const float* dec_w2  = (const float*)d_in[19];
    const float* dec_b2  = (const float*)d_in[20];
    const float* rew_w1  = (const float*)d_in[21];
    const float* rew_b1  = (const float*)d_in[22];
    const float* rew_w2  = (const float*)d_in[23];
    const float* rew_b2  = (const float*)d_in[24];
    float* out = (float*)d_out;

    // bump allocator on workspace (256B aligned)
    char*  ws  = (char*)d_ws;
    size_t off = 0;
    auto alloc = [&](size_t bytes) -> void* {
        off = (off + 255) & ~(size_t)255;
        void* p = ws + off;
        off += bytes;
        return p;
    };

    bf16* wt_enc1  = (bf16*)alloc((size_t)1024 * 512  * 2);
    bf16* wt_enc2  = (bf16*)alloc((size_t)1024 * 1024 * 2);
    bf16* wt_prior = (bf16*)alloc((size_t)512  * 1024 * 2);
    bf16* wt_post  = (bf16*)alloc((size_t)512  * 2048 * 2);
    bf16* wt_wih   = (bf16*)alloc((size_t)3072 * 288  * 2);
    bf16* wt_whh   = (bf16*)alloc((size_t)3072 * 1024 * 2);
    bf16* wt_dec1  = (bf16*)alloc((size_t)1024 * 1280 * 2);
    bf16* wt_dec2  = (bf16*)alloc((size_t)512  * 1024 * 2);
    bf16* wt_rew1  = (bf16*)alloc((size_t)1024 * 1280 * 2);

    bf16* obs_bf  = (bf16*)alloc((size_t)NROWS * OBS_D * 2);
    bf16* act_bf  = (bf16*)alloc((size_t)NROWS * ACT_D * 2);
    bf16* big1    = (bf16*)alloc((size_t)NROWS * 1024 * 2);   // e1 / d1 / r1 (reused)
    bf16* emb_bf  = (bf16*)alloc((size_t)NROWS * EMB_D * 2);
    bf16* hseq_bf = (bf16*)alloc((size_t)NROWS * DET_D * 2);
    bf16* zseq_bf = (bf16*)alloc((size_t)NROWS * STO_D * 2);

    float* pp      = (float*)alloc((size_t)BATCH * 512 * 4);
    float* qq      = (float*)alloc((size_t)BATCH * 512 * 4);
    float* gi_buf  = (float*)alloc((size_t)BATCH * 3072 * 4);
    float* gh_buf  = (float*)alloc((size_t)BATCH * 3072 * 4);
    float* h_f32   = (float*)alloc((size_t)BATCH * DET_D * 4);
    bf16*  h_bf    = (bf16*) alloc((size_t)BATCH * DET_D * 2);
    float* predobs = (float*)alloc((size_t)NROWS * OBS_D * 4);
    float* kl_buf    = (float*)alloc((size_t)NROWS * 4);
    float* recon_buf = (float*)alloc((size_t)NROWS * 4);
    float* rewl_buf  = (float*)alloc((size_t)NROWS * 4);

    // --- pack weights (W[KxN] -> WT[NxK] bf16; GRU weights already NxK) ---
    auto packT = [&](const float* W, bf16* WT, int K, int N) {
        int n = K * N;
        pack_transpose_kernel<<<(n + 255) / 256, 256, 0, stream>>>(W, WT, K, N);
    };
    packT(enc_w1, wt_enc1, 512, 1024);
    packT(enc_w2, wt_enc2, 1024, 1024);
    packT(prior_w, wt_prior, 1024, 512);
    packT(post_w,  wt_post,  2048, 512);
    pack_convert_kernel<<<(3072 * 288 + 255) / 256, 256, 0, stream>>>(gru_wih, wt_wih, 3072 * 288);
    pack_convert_kernel<<<(3072 * 1024 + 255) / 256, 256, 0, stream>>>(gru_whh, wt_whh, 3072 * 1024);
    packT(dec_w1, wt_dec1, 1280, 1024);
    packT(dec_w2, wt_dec2, 1024, 512);
    packT(rew_w1, wt_rew1, 1280, 1024);

    // --- convert activations, init h ---
    {
        int n = NROWS * OBS_D;
        f32_to_bf16_kernel<<<(n + 255) / 256, 256, 0, stream>>>(obs, obs_bf, n);
        n = NROWS * ACT_D;
        f32_to_bf16_kernel<<<(n + 255) / 256, 256, 0, stream>>>(actions, act_bf, n);
        zero_h_kernel<<<(BATCH * DET_D) / 256, 256, 0, stream>>>(h_f32, h_bf);
    }

    // --- batched encoder over all 16128 rows (h-independent) ---
    launch_gemm(stream, obs_bf, OBS_D, obs_bf, OBS_D, OBS_D, wt_enc1, enc_b1,
                nullptr, big1, NROWS, EMB_D, OBS_D, 1);
    launch_gemm(stream, big1, EMB_D, big1, EMB_D, EMB_D, wt_enc2, enc_b2,
                nullptr, emb_bf, NROWS, EMB_D, EMB_D, 1);

    // --- sequential scan: prior, posterior, GRU ---
    for (int t = 0; t < T_STEPS; ++t) {
        const bf16* emb_t = emb_bf + (size_t)t * BATCH * EMB_D;
        bf16* z_t = zseq_bf + (size_t)t * BATCH * STO_D;
        bf16* h_t = hseq_bf + (size_t)t * BATCH * DET_D;

        // prior: h @ prior_w
        launch_gemm(stream, h_bf, DET_D, h_bf, DET_D, DET_D, wt_prior, prior_b,
                    pp, nullptr, BATCH, 512, DET_D, 0);
        // post: [h | emb] @ post_w (split-A at k=1024)
        launch_gemm(stream, h_bf, DET_D, emb_t, EMB_D, DET_D, wt_post, post_b,
                    qq, nullptr, BATCH, 512, DET_D + EMB_D, 0);
        // z + KL
        latent_kl_kernel<<<BATCH, STO_D, 0, stream>>>(
            qq, pp, eps + (size_t)t * BATCH * STO_D, z_t, kl_buf + (size_t)t * BATCH);
        // gi = [z | a] @ Wih^T (split-A at k=256)
        launch_gemm(stream, z_t, STO_D, act_bf + (size_t)t * BATCH * ACT_D, ACT_D, STO_D,
                    wt_wih, gru_bih, gi_buf, nullptr, BATCH, 3072, STO_D + ACT_D, 0);
        // gh = h @ Whh^T
        launch_gemm(stream, h_bf, DET_D, h_bf, DET_D, DET_D, wt_whh, gru_bhh,
                    gh_buf, nullptr, BATCH, 3072, DET_D, 0);
        // gates + state update
        gru_update_kernel<<<(BATCH * DET_D) / 256, 256, 0, stream>>>(
            gi_buf, gh_buf, h_f32, h_bf, h_t, dones + (size_t)t * BATCH);
    }

    // --- batched heads over all 16128 rows: feat = [h_next | z] (split-A at 1024) ---
    launch_gemm(stream, hseq_bf, DET_D, zseq_bf, STO_D, DET_D, wt_dec1, dec_b1,
                nullptr, big1, NROWS, EMB_D, DET_D + STO_D, 1);
    launch_gemm(stream, big1, EMB_D, big1, EMB_D, EMB_D, wt_dec2, dec_b2,
                predobs, nullptr, NROWS, OBS_D, EMB_D, 0);
    recon_kernel<<<NROWS, 256, 0, stream>>>(predobs, obs, recon_buf);

    launch_gemm(stream, hseq_bf, DET_D, zseq_bf, STO_D, DET_D, wt_rew1, rew_b1,
                nullptr, big1, NROWS, EMB_D, DET_D + STO_D, 1);
    rew2_loss_kernel<<<NROWS, 256, 0, stream>>>(big1, rew_w2, rew_b2, rewards, rewl_buf);

    // --- final scalar reduction ---
    final_reduce_kernel<<<1, 256, 0, stream>>>(recon_buf, rewl_buf, kl_buf, out, NROWS);
}
